// TemporalViewModel_68968584839671
// MI455X (gfx1250) — matrix operand order repaired and checked by reading, not compiled, not run on hardware
//
#include <hip/hip_runtime.h>

// MI455X (gfx1250) GRU, fully pre-packed f16 pipeline:
//   pass 1a: met|ctx -> (T,N,32) f16 stream, spatial -> f16   (bandwidth-bound)
//   pass 1b: weights -> lane-indexed f16 WMMA B-fragments in ws (one-shot, tiny)
//   pass 2:  recurrence, 16 sequences per wave32, 12 v_wmma_f32_16x16x32_f16
//            per step; input-weight fragments in block LDS (off h critical path),
//            recurrent-weight fragments register-resident; h re-laid out C->A
//            each step through wave-private LDS (in-order per wave).

typedef __attribute__((ext_vector_type(16))) _Float16 v16h;
typedef __attribute__((ext_vector_type(8)))  _Float16 v8h;
typedef __attribute__((ext_vector_type(8)))  float    v8f;

#define WMMA_F16(a, b, c) \
  __builtin_amdgcn_wmma_f32_16x16x32_f16(false, (a), false, (b), (short)0, (c), false, false)

static constexpr int T_STEPS   = 24;
static constexpr int NITEMS    = 16384;   // B*X*Y = 16*32*32
static constexpr int FS        = 64;
static constexpr int FM        = 16;
static constexpr int FC        = 16;
static constexpr int ISIZE     = 96;      // FS+FM+FC
static constexpr int H         = 32;
static constexpr int WAVES     = 8;       // 256-thread block = 8 wave32
static constexpr int LDS_PITCH = 40;      // 32 halfs + 8 pad -> 80B rows, 16B aligned

// workspace layout (halfs unless noted)
static constexpr size_t XPK_OFF  = 0;                                  // (T,N,32)
static constexpr size_t SPPK_OFF = (size_t)T_STEPS * NITEMS * 32;      // (N,64)
static constexpr size_t WMC_OFF  = SPPK_OFF + (size_t)NITEMS * FS;     // [6][32][16]
static constexpr size_t WREC_OFF = WMC_OFF + 6 * 32 * 16;              // [6][32][16]
static constexpr size_t WSF_OFF  = WREC_OFF + 6 * 32 * 16;             // [2][6][32][16]
static constexpr size_t WOF_OFF  = WSF_OFF + 2 * 6 * 32 * 16;          // [4][32][16]
static constexpr size_t WS_HALFS = WOF_OFF + 4 * 32 * 16;

// ---------------- helpers ----------------

__device__ __forceinline__ v8h pack8(float4 a, float4 b) {
  v8h r;
  r[0] = (_Float16)a.x; r[1] = (_Float16)a.y; r[2] = (_Float16)a.z; r[3] = (_Float16)a.w;
  r[4] = (_Float16)b.x; r[5] = (_Float16)b.y; r[6] = (_Float16)b.z; r[7] = (_Float16)b.w;
  return r;
}

__device__ __forceinline__ v16h load_A_f16(const _Float16* lo, const _Float16* hi) {
  v8h a = *(const v8h*)lo;     // b128 each (global or ds)
  v8h b = *(const v8h*)hi;
  v16h r;
#pragma unroll
  for (int i = 0; i < 8; ++i) { r[i] = a[i]; r[8 + i] = b[i]; }
  return r;
}

__device__ __forceinline__ v8f bcast8(float x) {
  v8f r;
#pragma unroll
  for (int i = 0; i < 8; ++i) r[i] = x;
  return r;
}

__device__ __forceinline__ float sigmoid_f(float x) { return 1.0f / (1.0f + __expf(-x)); }
__device__ __forceinline__ float tanh_f(float x) {
  x = fminf(fmaxf(x, -15.0f), 15.0f);
  float e = __expf(-2.0f * x);
  return (1.0f - e) / (1.0f + e);
}

// ---------------- pass 1a: activations fp32 -> packed f16 ----------------
__global__ __launch_bounds__(256)
void TemporalViewModel_prepack(const float* __restrict__ met,
                               const float* __restrict__ ctx,
                               const float* __restrict__ spatial,
                               _Float16* __restrict__ xpk,
                               _Float16* __restrict__ sppk) {
  const int idx = blockIdx.x * blockDim.x + threadIdx.x;
  const int TN = T_STEPS * NITEMS;
  if (idx < TN) {
    const float4* m = (const float4*)(met + (size_t)idx * FM);
    const float4* c = (const float4*)(ctx + (size_t)idx * FC);
    v8h* o = (v8h*)(xpk + (size_t)idx * 32);
    o[0] = pack8(m[0], m[1]);
    o[1] = pack8(m[2], m[3]);
    o[2] = pack8(c[0], c[1]);
    o[3] = pack8(c[2], c[3]);
  } else if (idx < TN + NITEMS) {
    const int it = idx - TN;
    const float4* s = (const float4*)(spatial + (size_t)it * FS);
    v8h* o = (v8h*)(sppk + (size_t)it * FS);
#pragma unroll
    for (int q = 0; q < 8; ++q) o[q] = pack8(s[2 * q], s[2 * q + 1]);
  }
}

// ---------------- pass 1b: weights -> lane-indexed f16 B fragments ----------------
// fragment element (tile j, lane, e): colg = j*16 + (lane&15); k = (lane>=16)*8 + (e&7) + (e>=8)*16
__global__ __launch_bounds__(256)
void TemporalViewModel_wprep(const float* __restrict__ W_ih,
                             const float* __restrict__ W_hh,
                             const float* __restrict__ Wout,
                             _Float16* __restrict__ wmc,   // [6][32][16]
                             _Float16* __restrict__ wrec,  // [6][32][16]
                             _Float16* __restrict__ wsf,   // [2][6][32][16]
                             _Float16* __restrict__ wof) { // [4][32][16]
  int i = blockIdx.x * blockDim.x + threadIdx.x;
  if (i < 6 * 32 * 16) {                       // wmc: W_ih^T met|ctx rows (64..95)
    const int e = i & 15, lane = (i >> 4) & 31, j = i >> 9;
    const int colg = j * 16 + (lane & 15), kb = (lane >> 4) * 8;
    wmc[i] = (_Float16)W_ih[colg * ISIZE + FS + kb + (e & 7) + ((e >> 3) << 4)];
    return;
  }
  i -= 6 * 32 * 16;
  if (i < 6 * 32 * 16) {                       // wrec: W_hh^T
    const int e = i & 15, lane = (i >> 4) & 31, j = i >> 9;
    const int colg = j * 16 + (lane & 15), kb = (lane >> 4) * 8;
    wrec[i] = (_Float16)W_hh[colg * H + kb + (e & 7) + ((e >> 3) << 4)];
    return;
  }
  i -= 6 * 32 * 16;
  if (i < 2 * 6 * 32 * 16) {                   // wsf: W_ih^T spatial rows (kc*32..)
    const int e = i & 15, lane = (i >> 4) & 31, j = (i >> 9) % 6, kc = i >> 12;
    const int colg = j * 16 + (lane & 15), kb = (lane >> 4) * 8;
    wsf[i] = (_Float16)W_ih[colg * ISIZE + kc * 32 + kb + (e & 7) + ((e >> 3) << 4)];
    return;
  }
  i -= 2 * 6 * 32 * 16;
  if (i < 4 * 32 * 16) {                       // wof: Wout (k strided by FS)
    const int e = i & 15, lane = (i >> 4) & 31, j = i >> 9;
    const int colo = j * 16 + (lane & 15), kb = (lane >> 4) * 8;
    wof[i] = (_Float16)Wout[(kb + (e & 7) + ((e >> 3) << 4)) * FS + colo];
  }
}

// ---------------- pass 2: GRU recurrence ----------------

__global__ __launch_bounds__(256)
void TemporalViewModel_68968584839671_kernel(
    const _Float16* __restrict__ xpk,    // (T, N, 32) f16 (met|ctx)
    const _Float16* __restrict__ sppk,   // (N, 64) f16
    const _Float16* __restrict__ wmc,    // [6][32][16] f16 fragments
    const _Float16* __restrict__ wrec,   // [6][32][16]
    const _Float16* __restrict__ wsf,    // [2][6][32][16]
    const _Float16* __restrict__ wof,    // [4][32][16]
    const float* __restrict__ b_ih,      // (96,)
    const float* __restrict__ b_hh,      // (96,)
    float* __restrict__ out)             // (N, 64)
{
  __shared__ __align__(16) _Float16 wlds[6][32][16];       // input-weight fragments
  __shared__ __align__(16) _Float16 hlds[WAVES][16][LDS_PITCH];  // h transpose staging

  const int lane  = threadIdx.x & 31;
  const int wave  = threadIdx.x >> 5;
  const int half  = lane >> 4;
  const int ln    = lane & 15;
  const int khalf = half * 8;        // A-layout K base
  const int itemBase = (blockIdx.x * WAVES + wave) * 16;
  const int item     = itemBase + ln;

  // ---- wave 0 copies W_mc fragments to LDS (raw f16, b128 in/out) ----
  if (wave == 0) {
#pragma unroll
    for (int j = 0; j < 6; ++j) {
      const _Float16* src = wmc + (j * 32 + lane) * 16;
      *(v8h*)&wlds[j][lane][0] = *(const v8h*)src;
      *(v8h*)&wlds[j][lane][8] = *(const v8h*)(src + 8);
    }
  }
  // ---- recurrent-weight fragments register-resident (no staging, no cvts) ----
  v16h w_rec[6];
#pragma unroll
  for (int j = 0; j < 6; ++j) {
    const _Float16* src = wrec + (j * 32 + lane) * 16;
    w_rec[j] = load_A_f16(src, src + 8);
  }
  __syncthreads();

  // ---- per-sequence constant: gis = b + spatial @ W_s^T (hoisted out of t) ----
  v8f gis[6];
#pragma unroll
  for (int j = 0; j < 6; ++j) {
    const int colg = j * 16 + ln;
    gis[j] = bcast8(b_ih[colg] + ((j < 4) ? b_hh[colg] : 0.0f));
  }
  const float bhn0 = b_hh[2 * H + ln];        // b_hh n-part enters via VALU below
  const float bhn1 = b_hh[2 * H + 16 + ln];

#pragma unroll
  for (int kc = 0; kc < 2; ++kc) {
    const _Float16* sp = sppk + (size_t)item * FS + kc * 32 + khalf;
    v16h a_s = load_A_f16(sp, sp + 16);
#pragma unroll
    for (int j = 0; j < 6; ++j) {
      const _Float16* src = wsf + ((kc * 6 + j) * 32 + lane) * 16;
      v16h w_s = load_A_f16(src, src + 8);
      gis[j] = WMMA_F16(a_s, w_s, gis[j]);
    }
  }

  // ---- recurrence ----
  v8f h0 = bcast8(0.0f), h1 = bcast8(0.0f);
  _Float16* myrow = &hlds[wave][ln][0];

  for (int t = 0; t < T_STEPS; ++t) {
    // spill h (C layout) -> LDS f16; reload as A tile (wave-private, in-order LDS)
#pragma unroll
    for (int v = 0; v < 8; ++v) {
      const int r = half * 8 + v;
      hlds[wave][r][ln]      = (_Float16)h0[v];
      hlds[wave][r][16 + ln] = (_Float16)h1[v];
    }
    v16h h_a = load_A_f16(myrow + khalf, myrow + 16 + khalf);

    // packed met|ctx step tile: two global_load_b128
    const size_t tb = (size_t)t * NITEMS;
    const _Float16* xp = xpk + (tb + item) * 32 + khalf;
    v16h x_mc = load_A_f16(xp, xp + 16);
    if (t + 1 < T_STEPS) __builtin_prefetch(xpk + (tb + NITEMS + item) * 32, 0, 1);

    // r,z tiles 0..3: input part (B from LDS) then recurrent part (B resident)
    v8f g0, g1, g2, g3;
    {
      v16h wm = load_A_f16(&wlds[0][lane][0], &wlds[0][lane][8]);
      g0 = WMMA_F16(x_mc, wm, gis[0]);
    }
    {
      v16h wm = load_A_f16(&wlds[1][lane][0], &wlds[1][lane][8]);
      g1 = WMMA_F16(x_mc, wm, gis[1]);
    }
    {
      v16h wm = load_A_f16(&wlds[2][lane][0], &wlds[2][lane][8]);
      g2 = WMMA_F16(x_mc, wm, gis[2]);
    }
    {
      v16h wm = load_A_f16(&wlds[3][lane][0], &wlds[3][lane][8]);
      g3 = WMMA_F16(x_mc, wm, gis[3]);
    }
    g0 = WMMA_F16(h_a, w_rec[0], g0);
    g1 = WMMA_F16(h_a, w_rec[1], g1);
    g2 = WMMA_F16(h_a, w_rec[2], g2);
    g3 = WMMA_F16(h_a, w_rec[3], g3);

    // n tiles 4,5: input part (with gis C) and recurrent part (inline-0 C)
    v8f gin0, gin1;
    {
      v16h wm = load_A_f16(&wlds[4][lane][0], &wlds[4][lane][8]);
      gin0 = WMMA_F16(x_mc, wm, gis[4]);
    }
    {
      v16h wm = load_A_f16(&wlds[5][lane][0], &wlds[5][lane][8]);
      gin1 = WMMA_F16(x_mc, wm, gis[5]);
    }
    v8f zc = {};  // lowers to inline SRC2=0
    v8f ghn0 = WMMA_F16(h_a, w_rec[4], zc);
    v8f ghn1 = WMMA_F16(h_a, w_rec[5], zc);

    // gate math + state update (lane-local VALU); b_hn applied here
#pragma unroll
    for (int v = 0; v < 8; ++v) {
      float r0 = sigmoid_f(g0[v]);
      float r1 = sigmoid_f(g1[v]);
      float z0 = sigmoid_f(g2[v]);
      float z1 = sigmoid_f(g3[v]);
      float n0 = tanh_f(gin0[v] + r0 * (ghn0[v] + bhn0));
      float n1 = tanh_f(gin1[v] + r1 * (ghn1[v] + bhn1));
      h0[v] = n0 + z0 * (h0[v] - n0);
      h1[v] = n1 + z1 * (h1[v] - n1);
    }
  }

  // ---- output projection: out = hT @ W (32x64) ----
#pragma unroll
  for (int v = 0; v < 8; ++v) {
    const int r = half * 8 + v;
    hlds[wave][r][ln]      = (_Float16)h0[v];
    hlds[wave][r][16 + ln] = (_Float16)h1[v];
  }
  v16h h_a = load_A_f16(myrow + khalf, myrow + 16 + khalf);

#pragma unroll
  for (int j = 0; j < 4; ++j) {
    const int colo = j * 16 + ln;
    const _Float16* src = wof + (j * 32 + lane) * 16;
    v16h wb = load_A_f16(src, src + 8);
    v8f zc = {};
    v8f o = WMMA_F16(h_a, wb, zc);
#pragma unroll
    for (int v = 0; v < 8; ++v) {
      const int r = half * 8 + v;
      out[(size_t)(itemBase + r) * FS + colo] = o[v];
    }
  }
}

// ---------------- launch ----------------

extern "C" void kernel_launch(void* const* d_in, const int* in_sizes, int n_in,
                              void* d_out, int out_size, void* d_ws, size_t ws_size,
                              hipStream_t stream) {
  const float* spatial = (const float*)d_in[0];
  const float* met     = (const float*)d_in[1];
  const float* ctx     = (const float*)d_in[2];
  const float* W_ih    = (const float*)d_in[3];
  const float* W_hh    = (const float*)d_in[4];
  const float* b_ih    = (const float*)d_in[5];
  const float* b_hh    = (const float*)d_in[6];
  const float* Wout    = (const float*)d_in[7];
  float* out = (float*)d_out;

  _Float16* ws   = (_Float16*)d_ws;          // ~27.3 MB of f16 scratch
  _Float16* xpk  = ws + XPK_OFF;
  _Float16* sppk = ws + SPPK_OFF;
  _Float16* wmc  = ws + WMC_OFF;
  _Float16* wrec = ws + WREC_OFF;
  _Float16* wsf  = ws + WSF_OFF;
  _Float16* wof  = ws + WOF_OFF;

  const int totalA = T_STEPS * NITEMS + NITEMS;
  TemporalViewModel_prepack<<<(totalA + 255) / 256, 256, 0, stream>>>(
      met, ctx, spatial, xpk, sppk);

  const int totalW = 6 * 32 * 16 + 6 * 32 * 16 + 2 * 6 * 32 * 16 + 4 * 32 * 16;
  TemporalViewModel_wprep<<<(totalW + 255) / 256, 256, 0, stream>>>(
      W_ih, W_hh, Wout, wmc, wrec, wsf, wof);

  dim3 grid(NITEMS / (WAVES * 16));   // 128 blocks, 8 wave32 each, 16 seqs/wave
  dim3 block(WAVES * 32);             // 256 threads
  TemporalViewModel_68968584839671_kernel<<<grid, block, 0, stream>>>(
      xpk, sppk, wmc, wrec, wsf, wof, b_ih, b_hh, out);
}
